// Quantisation_21620865368396
// MI455X (gfx1250) — matrix operand order repaired and checked by reading, not compiled, not run on hardware
//
#include <hip/hip_runtime.h>
#include <hip/hip_bf16.h>

#define N_PTS  32768
#define K_CODE 8192
#define D_DIM  256

// LDS staging geometry: 16 codebook rows x 512B, padded to 528B/row so the
// 16 per-lane streams start on distinct banks (4*l15 mod 64).
#define ROW_PITCH  528
#define TILE_BYTES (16 * ROW_PITCH)     // 8448 B per matrix (hi or lo)
#define BUF_BYTES  (2 * TILE_BYTES)     // hi + lo

typedef __attribute__((ext_vector_type(16))) __bf16 v16bf;
typedef __attribute__((ext_vector_type(8)))  __bf16 v8bf;
typedef __attribute__((ext_vector_type(8)))  float  v8f;

// ---- WMMA fragment loaders (bf16, wave32) -------------------------------
// A 16x32 (MxK): lanes 0-15 hold row M=lane, K = {d0..d0+7, d0+16..d0+23};
//                lanes 16-31 hold row M=lane-16, K = {d0+8.., d0+24..}.
static __device__ inline v16bf load_frag_a(const __bf16* p) {
    union { v16bf v; uint4 q[2]; } u;
    u.q[0] = *(const uint4*)(p);
    u.q[1] = *(const uint4*)(p + 16);
    return u.v;
}
// B 32x16 (KxN) fragment from LDS: two 16-byte chunks of a staged W row, 32 bytes apart.
static __device__ inline v16bf load_frag_b_lds(const char* p) {
    union { v16bf v; uint4 q[2]; } u;
    u.q[0] = *(const uint4*)(p);
    u.q[1] = *(const uint4*)(p + 32);
    return u.v;
}

// ---- 1) split x into bf16 hi + bf16 residual ----------------------------
__global__ __launch_bounds__(256) void vq_prep_x(const float* __restrict__ x,
                                                 __bf16* __restrict__ xh,
                                                 __bf16* __restrict__ xl) {
    int gid = blockIdx.x * 256 + threadIdx.x;          // 8 floats per thread
    const float4* px = (const float4*)x + (size_t)gid * 2;
    float4 f0 = px[0], f1 = px[1];
    float v[8] = {f0.x, f0.y, f0.z, f0.w, f1.x, f1.y, f1.z, f1.w};
    v8bf h, l;
#pragma unroll
    for (int i = 0; i < 8; ++i) {
        __bf16 hb = (__bf16)v[i];
        h[i] = hb;
        l[i] = (__bf16)(v[i] - (float)hb);
    }
    *(v8bf*)(xh + (size_t)gid * 8) = h;
    *(v8bf*)(xl + (size_t)gid * 8) = l;
}

// ---- 2) split w + per-row ||w||^2 (one wave per codebook row) -----------
__global__ __launch_bounds__(256) void vq_prep_w(const float* __restrict__ w,
                                                 __bf16* __restrict__ wh,
                                                 __bf16* __restrict__ wl,
                                                 float* __restrict__ w2) {
    int wave = threadIdx.x >> 5, lane = threadIdx.x & 31;
    int row = blockIdx.x * 8 + wave;
    const float4* p = (const float4*)(w + (size_t)row * D_DIM + lane * 8);
    float4 f0 = p[0], f1 = p[1];
    float v[8] = {f0.x, f0.y, f0.z, f0.w, f1.x, f1.y, f1.z, f1.w};
    float s = 0.f;
    v8bf h, l;
#pragma unroll
    for (int i = 0; i < 8; ++i) {
        __bf16 hb = (__bf16)v[i];
        h[i] = hb;
        l[i] = (__bf16)(v[i] - (float)hb);
        s = fmaf(v[i], v[i], s);
    }
    *(v8bf*)(wh + (size_t)row * D_DIM + lane * 8) = h;
    *(v8bf*)(wl + (size_t)row * D_DIM + lane * 8) = l;
#pragma unroll
    for (int off = 16; off >= 1; off >>= 1) s += __shfl_xor(s, off, 32);
    if (lane == 0) w2[row] = s;
}

// ---- 3) fused split-bf16 GEMM + running argmin --------------------------
// Each wave owns TWO 16-row A tiles (32 x-rows) in registers for all of
// D=256 (hi+lo = 256 VGPRs). The 8 waves cooperatively DMA each 16-col
// B tile (hi+lo, 16 KB) into double-buffered LDS with
// GLOBAL_LOAD_ASYNC_TO_LDS_B128, overlapped with the 48 WMMAs on the
// previous tile; fragments are then read via ds_load_b128.
__global__ __launch_bounds__(256) void vq_gemm_argmin(
    const __bf16* __restrict__ xh, const __bf16* __restrict__ xl,
    const __bf16* __restrict__ wh, const __bf16* __restrict__ wl,
    const float* __restrict__ w2, int* __restrict__ amin) {
    __shared__ __align__(16) char smem[2][BUF_BYTES];

    const int tid  = threadIdx.x;
    const int wave = tid >> 5;
    const int lane = tid & 31;
    const int hf   = lane >> 4;      // half-wave select
    const int l15  = lane & 15;
    const int rowBase = blockIdx.x * 256 + wave * 32;

    // --- cooperative async stage of one column tile (hi+lo) into smem[db]
    // 1024 x 16B chunks, 4 per thread. ASYNCcnt-tracked.
    auto stage_tile = [&](int colBase, int db) {
#pragma unroll
        for (int i = 0; i < 4; ++i) {
            int id  = tid + 256 * i;         // 0..1023
            int m   = id >> 9;               // 0 = hi, 1 = lo
            int rc  = id & 511;
            int row = rc >> 5;               // 0..15
            int c16 = rc & 31;               // 16B chunk within 512B row
            const __bf16* g = (m ? wl : wh) +
                              (size_t)(colBase + row) * D_DIM + c16 * 8;
            unsigned lds = (unsigned)(uintptr_t)(&smem[db][0] +
                            m * TILE_BYTES + row * ROW_PITCH + c16 * 16);
            asm volatile("global_load_async_to_lds_b128 %0, %1, off"
                         :: "v"(lds), "v"(g) : "memory");
        }
    };

    // --- preload A (hi+lo) for 32 rows across all of D -------------------
    v16bf ah[2][8], al[2][8];
#pragma unroll
    for (int t = 0; t < 2; ++t) {
        const __bf16* pah = xh + (size_t)(rowBase + 16 * t + l15) * D_DIM + hf * 8;
        const __bf16* pal = xl + (size_t)(rowBase + 16 * t + l15) * D_DIM + hf * 8;
#pragma unroll
        for (int ks = 0; ks < 8; ++ks) {
            ah[t][ks] = load_frag_a(pah + ks * 32);
            al[t][ks] = load_frag_a(pal + ks * 32);
        }
    }

    float minv[2][8];
    int   mini[2][8];
#pragma unroll
    for (int t = 0; t < 2; ++t)
#pragma unroll
        for (int r = 0; r < 8; ++r) { minv[t][r] = __builtin_huge_valf(); mini[t][r] = 0; }

    // --- pipeline prologue: stage tile 0 ---------------------------------
    stage_tile(0, 0);
    asm volatile("s_wait_asynccnt 0x0" ::: "memory");
    __syncthreads();

    // --- stream all 512 column tiles of the codebook ---------------------
    for (int colBase = 0; colBase < K_CODE; colBase += 16) {
        const int db = (colBase >> 4) & 1;
        if (colBase + 16 < K_CODE) stage_tile(colBase + 16, db ^ 1);

        // per-lane B fragment base inside the staged tile
        const char* bh0 = &smem[db][0] + l15 * ROW_PITCH + hf * 16;
        const char* bl0 = bh0 + TILE_BYTES;

        v8f acc0 = {}, acc1 = {};
#pragma unroll
        for (int ks = 0; ks < 8; ++ks) {
            v16bf bh = load_frag_b_lds(bh0 + ks * 64);
            v16bf bl = load_frag_b_lds(bl0 + ks * 64);
            acc0 = __builtin_amdgcn_wmma_f32_16x16x32_bf16(
                false, ah[0][ks], false, bh, (short)0, acc0, false, false);
            acc1 = __builtin_amdgcn_wmma_f32_16x16x32_bf16(
                false, ah[1][ks], false, bh, (short)0, acc1, false, false);
            acc0 = __builtin_amdgcn_wmma_f32_16x16x32_bf16(
                false, ah[0][ks], false, bl, (short)0, acc0, false, false);
            acc1 = __builtin_amdgcn_wmma_f32_16x16x32_bf16(
                false, ah[1][ks], false, bl, (short)0, acc1, false, false);
            acc0 = __builtin_amdgcn_wmma_f32_16x16x32_bf16(
                false, al[0][ks], false, bh, (short)0, acc0, false, false);
            acc1 = __builtin_amdgcn_wmma_f32_16x16x32_bf16(
                false, al[1][ks], false, bh, (short)0, acc1, false, false);
        }
        // score = ||w_c||^2 - 2 * x.w_c  (||x||^2 is constant per row)
        const int   c   = colBase + l15;
        const float w2c = w2[c];
#pragma unroll
        for (int r = 0; r < 8; ++r) {
            float s0 = fmaf(-2.0f, acc0[r], w2c);
            bool  t0 = s0 < minv[0][r];
            minv[0][r] = t0 ? s0 : minv[0][r];
            mini[0][r] = t0 ? c  : mini[0][r];
            float s1 = fmaf(-2.0f, acc1[r], w2c);
            bool  t1 = s1 < minv[1][r];
            minv[1][r] = t1 ? s1 : minv[1][r];
            mini[1][r] = t1 ? c  : mini[1][r];
        }

        // own async stages done, then make them visible workgroup-wide;
        // also guarantees nobody still reads the buffer we write next.
        asm volatile("s_wait_asynccnt 0x0" ::: "memory");
        __syncthreads();
    }

    // Cross-lane (width-16) min+index reduction; tie -> lowest index.
#pragma unroll
    for (int t = 0; t < 2; ++t)
#pragma unroll
        for (int r = 0; r < 8; ++r) {
            float v = minv[t][r];
            int   i = mini[t][r];
#pragma unroll
            for (int off = 8; off >= 1; off >>= 1) {
                float ov = __shfl_xor(v, off, 32);
                int   oi = __shfl_xor(i, off, 32);
                if (ov < v || (ov == v && oi < i)) { v = ov; i = oi; }
            }
            if (l15 == 0) amin[rowBase + 16 * t + hf * 8 + r] = i;  // M = r + 8*hf
        }
}

// ---- 4) gather: out[n,:] = fp32 weight[argmin[n],:] ---------------------
__global__ __launch_bounds__(256) void vq_gather(const float* __restrict__ w,
                                                 const int* __restrict__ amin,
                                                 float* __restrict__ out) {
    int gid = blockIdx.x * 256 + threadIdx.x;          // one float4 each
    int n   = gid >> 6;                                // D/4 = 64 float4/row
    int c4  = gid & 63;
    int k   = amin[n];
    ((float4*)out)[(size_t)n * 64 + c4] = ((const float4*)w)[(size_t)k * 64 + c4];
}

// ---- launcher -----------------------------------------------------------
extern "C" void kernel_launch(void* const* d_in, const int* in_sizes, int n_in,
                              void* d_out, int out_size, void* d_ws, size_t ws_size,
                              hipStream_t stream) {
    (void)in_sizes; (void)n_in; (void)out_size; (void)ws_size;
    const float* x = (const float*)d_in[0];
    const float* w = (const float*)d_in[1];
    float* out = (float*)d_out;

    char* ws = (char*)d_ws;
    size_t off = 0;
    auto take = [&](size_t bytes) {
        void* p = ws + off;
        off = (off + bytes + 255) & ~(size_t)255;
        return p;
    };
    __bf16* xh  = (__bf16*)take((size_t)N_PTS * D_DIM * sizeof(__bf16));
    __bf16* xl  = (__bf16*)take((size_t)N_PTS * D_DIM * sizeof(__bf16));
    __bf16* wh  = (__bf16*)take((size_t)K_CODE * D_DIM * sizeof(__bf16));
    __bf16* wl  = (__bf16*)take((size_t)K_CODE * D_DIM * sizeof(__bf16));
    float*  w2  = (float*)take((size_t)K_CODE * sizeof(float));
    int*    ami = (int*)take((size_t)N_PTS * sizeof(int));

    vq_prep_x<<<(N_PTS * D_DIM / 8) / 256, 256, 0, stream>>>(x, xh, xl);
    vq_prep_w<<<K_CODE / 8, 256, 0, stream>>>(w, wh, wl, w2);
    vq_gemm_argmin<<<N_PTS / 256, 256, 0, stream>>>(xh, xl, wh, wl, w2, ami);
    vq_gather<<<(N_PTS * 64) / 256, 256, 0, stream>>>(w, ami, out);
}
